// AttenMod_82068235092067
// MI455X (gfx1250) — compile-verified, matrix operand order checked by reading, not run on hardware
//
#include <hip/hip_runtime.h>
#include <hip/hip_bf16.h>

// ============================================================================
// MI455X (gfx1250) implementation.
//
// All large GEMMs run on V_WMMA_F32_16X16X32_F16 with fp32 accumulation.
// fp32 operands are pre-packed into the ISA fragment layout (CDNA5 ISA
// 7.12.2): per lane, 16 contiguous f16 covering k = e + 8*(lane/16) +
// (e>=8 ? 8 : 0), row/col = lane%16.
//
// GEMM kernel: 64x32 C macro-tile per wave (4x2 grid of 16x16 WMMA tiles,
// 8 v8f accumulators = 64 VGPRs; 6 fragments = 48 VGPRs -> no scratch spills
// under a waves_per_eu(2) register budget). Per k-step: 4 A-frags + 2 B-frags
// (6 KB of contiguous 32B-per-lane loads -> global_load_b128 clauses) feed
// 8 WMMAs (~21 flop/B reuse), with unguarded speculative global_prefetch_b8
// lookahead on all 6 streams. All M are multiples of 64 and N multiples of 32
// in this model, so no edge handling and EXEC stays all-ones.
// Convs, softmax, layernorm are bandwidth-bound VALU kernels.
// ============================================================================

typedef __attribute__((ext_vector_type(16))) _Float16 v16h;
typedef __attribute__((ext_vector_type(8)))  float    v8f;

#define WAVES_PER_BLOCK 8

// ---------------------------------------------------------------------------
// Pack fp32 row-major X[R,K] into WMMA fragment-layout f16 tiles.
// Tile (rt,kt) = 32 lanes x 16 f16 (1024 B). Zero-fills K padding (e.g. K=36->64).
// ---------------------------------------------------------------------------
__global__ void pack_f16_kernel(const float* __restrict__ X, _Float16* __restrict__ P,
                                long R, int K) {
  int  nKT = (K + 31) >> 5;
  long nRT = (R + 15) >> 4;
  long idx = (long)blockIdx.x * blockDim.x + threadIdx.x;
  long total = nRT * (long)nKT * 32;
  if (idx >= total) return;
  int  lane = (int)(idx & 31);
  long t    = idx >> 5;
  int  kt   = (int)(t % nKT);
  long rt   = t / nKT;
  _Float16* Pp = P + ((rt * nKT + kt) * 32 + lane) * 16;
  long r  = rt * 16 + (lane & 15);
  int  hl = lane >> 4;
#pragma unroll
  for (int e = 0; e < 16; ++e) {
    int k = kt * 32 + e + 8 * hl + ((e >= 8) ? 8 : 0);
    float v = (r < R && k < K) ? X[r * (long)K + k] : 0.0f;
    Pp[e] = (_Float16)v;
  }
}

// ---------------------------------------------------------------------------
// Batched WMMA GEMM: C[b][m,n] = sum_k A[b][m,k] * W[b][n,k] + bias[b][n]
// A,W pre-packed fragments. 64x32 C macro-tile per wave (4x2 WMMA tiles).
// Requires M%64==0, N%32==0 (true for every call site in this model).
// ---------------------------------------------------------------------------
__global__ void __attribute__((amdgpu_waves_per_eu(2)))
gemm_wmma_f16(const _Float16* __restrict__ Ap, const _Float16* __restrict__ Wp,
              const float* __restrict__ bias, float* __restrict__ C,
              int M, int N, int K,
              long aBS, long wBS, long bBS, long cBS, int relu) {
  int nNT = N >> 5;                       // 32-wide macro-tiles (2 x 16)
  int nMT = M >> 6;                       // 64-tall macro-tiles (4 x 16)
  int nKT = (K + 31) >> 5;
  int wave = threadIdx.x >> 5;
  int lane = threadIdx.x & 31;
  long tile = (long)blockIdx.x * WAVES_PER_BLOCK + wave;
  long ntiles = (long)nMT * nNT;
  if (tile >= ntiles) return;             // wave-uniform: EXEC stays all-ones for WMMA
  int tm = (int)(tile / nNT);
  int tn = (int)(tile % nNT);
  long batch = blockIdx.y;
  long rowStride = (long)nKT * 512;       // f16 elements between consecutive 16-row tiles
  const _Float16* A = Ap + batch * aBS + ((long)tm * 4) * rowStride + (long)lane * 16;
  const _Float16* W = Wp + batch * wBS + ((long)tn * 2) * rowStride + (long)lane * 16;

  v8f acc[4][2] = {};
  for (int kt = 0; kt < nKT; ++kt) {
    long ko = (long)kt * 512;
    v16h a[4], b[2];
#pragma unroll
    for (int i = 0; i < 4; ++i) a[i] = *(const v16h*)(A + i * rowStride + ko);
#pragma unroll
    for (int j = 0; j < 2; ++j) b[j] = *(const v16h*)(W + j * rowStride + ko);
    // speculative prefetch of next 1KB tile on each stream (safe past the end)
    {
      long kp = ko + 512;
#pragma unroll
      for (int i = 0; i < 4; ++i)
        __builtin_prefetch((const void*)(A + i * rowStride + kp), 0, 0);
#pragma unroll
      for (int j = 0; j < 2; ++j)
        __builtin_prefetch((const void*)(W + j * rowStride + kp), 0, 0);
    }
#pragma unroll
    for (int i = 0; i < 4; ++i)
#pragma unroll
      for (int j = 0; j < 2; ++j)
        acc[i][j] = __builtin_amdgcn_wmma_f32_16x16x32_f16(
            /*neg_a=*/false, a[i], /*neg_b=*/false, b[j],
            /*c_mod=*/(short)0, acc[i][j], /*reuse_a=*/false, /*reuse_b=*/false);
  }

  int hl = lane >> 4;
  int nl = lane & 15;
  float* Cb = C + batch * cBS;
#pragma unroll
  for (int j = 0; j < 2; ++j) {
    int n = tn * 32 + j * 16 + nl;
    float bv = bias ? bias[batch * bBS + n] : 0.0f;
#pragma unroll
    for (int i = 0; i < 4; ++i) {
#pragma unroll
      for (int r2 = 0; r2 < 8; ++r2) {
        long m = (long)tm * 64 + i * 16 + r2 + 8 * hl;  // C/D: M = vgpr + 8*half, N = lane%16
        float v = acc[i][j][r2] + bv;
        if (relu) v = fmaxf(v, 0.0f);
        Cb[m * N + n] = v;
      }
    }
  }
}

// ---------------------------------------------------------------------------
// conv1 (1->32ch, 3x3 valid on 32x32 tiles of t/255) fused with 2x2 maxpool.
// out: (bs*16, 32, 15, 15)
// ---------------------------------------------------------------------------
__global__ void conv1_pool_kernel(const float* __restrict__ t, const float* __restrict__ w,
                                  const float* __restrict__ b, float* __restrict__ out, int bs) {
  long total = (long)bs * 16 * 32 * 15 * 15;
  long idx = (long)blockIdx.x * blockDim.x + threadIdx.x;
  if (idx >= total) return;
  int ox = (int)(idx % 15);
  int oy = (int)((idx / 15) % 15);
  int oc = (int)((idx / 225) % 32);
  long n  = idx / (225 * 32);
  int tile = (int)(n & 15);
  long bimg = n >> 4;
  int ty = tile >> 2, tx = tile & 3;
  const float* img = t + (bimg * 128 + (long)ty * 32) * 128 + tx * 32;
  const float* wk = w + oc * 9;
  const float inv255 = 1.0f / 255.0f;
  float m = -3.4e38f;
#pragma unroll
  for (int py = 0; py < 2; ++py)
#pragma unroll
    for (int px = 0; px < 2; ++px) {
      int cy = oy * 2 + py, cx = ox * 2 + px;
      float s = 0.0f;
#pragma unroll
      for (int ky = 0; ky < 3; ++ky)
#pragma unroll
        for (int kx = 0; kx < 3; ++kx)
          s += img[(long)(cy + ky) * 128 + cx + kx] * wk[ky * 3 + kx];
      m = fmaxf(m, s * inv255 + b[oc]);
    }
  out[idx] = m;
}

// ---------------------------------------------------------------------------
// conv2 (32->32ch, 3x3 valid on 15x15) fused with 2x2 maxpool. out: (N,32,6,6)
// ---------------------------------------------------------------------------
__global__ void conv2_pool_kernel(const float* __restrict__ u1, const float* __restrict__ w,
                                  const float* __restrict__ b, float* __restrict__ out, long nimg) {
  long total = nimg * 32 * 36;
  long idx = (long)blockIdx.x * blockDim.x + threadIdx.x;
  if (idx >= total) return;
  int ox = (int)(idx % 6);
  int oy = (int)((idx / 6) % 6);
  int oc = (int)((idx / 36) % 32);
  long n  = idx / (36 * 32);
  const float* in0 = u1 + n * 32 * 225;
  float m = -3.4e38f;
#pragma unroll
  for (int py = 0; py < 2; ++py)
#pragma unroll
    for (int px = 0; px < 2; ++px) {
      int cy = oy * 2 + py, cx = ox * 2 + px;
      float s = b[oc];
      for (int ic = 0; ic < 32; ++ic) {
        const float* ip = in0 + ic * 225 + cy * 15 + cx;
        const float* wk = w + ((long)oc * 32 + ic) * 9;
#pragma unroll
        for (int ky = 0; ky < 3; ++ky)
#pragma unroll
          for (int kx = 0; kx < 3; ++kx)
            s += ip[ky * 15 + kx] * wk[ky * 3 + kx];
      }
      m = fmaxf(m, s);
    }
  out[idx] = m;
}

// ---------------------------------------------------------------------------
// Grouped MHA core: G=16, L=32, heads=4, hd=16. qkv: (16, 32*bs, 192) fp32.
// One block per (g,b,h), one thread per query row l. Writes pre-Wo output.
// ---------------------------------------------------------------------------
__global__ void grouped_attn_kernel(const float* __restrict__ qkv, float* __restrict__ o, int bs) {
  int blk = blockIdx.x;
  int h = blk & 3;
  int b = (blk >> 2) % bs;
  int g = blk / (4 * bs);
  int l = threadIdx.x;  // 0..31
  long rowsPerG = (long)32 * bs;
  const float* base = qkv + (long)g * rowsPerG * 192;
  float q[16];
  const float* qr = base + ((long)l * bs + b) * 192 + h * 16;
#pragma unroll
  for (int d = 0; d < 16; ++d) q[d] = qr[d] * 0.25f;   // hd^-0.5 = 1/4
  float sc[32];
  float mx = -3.4e38f;
#pragma unroll
  for (int kl = 0; kl < 32; ++kl) {
    const float* kr = base + ((long)kl * bs + b) * 192 + 64 + h * 16;
    float s = 0.0f;
#pragma unroll
    for (int d = 0; d < 16; ++d) s += q[d] * kr[d];
    sc[kl] = s;
    mx = fmaxf(mx, s);
  }
  float sum = 0.0f;
#pragma unroll
  for (int kl = 0; kl < 32; ++kl) { float e = __expf(sc[kl] - mx); sc[kl] = e; sum += e; }
  float inv = 1.0f / sum;
  float oa[16] = {};
#pragma unroll
  for (int kl = 0; kl < 32; ++kl) {
    float wgt = sc[kl] * inv;
    const float* vr = base + ((long)kl * bs + b) * 192 + 128 + h * 16;
#pragma unroll
    for (int d = 0; d < 16; ++d) oa[d] += wgt * vr[d];
  }
  float* orow = o + ((long)g * rowsPerG + (long)l * bs + b) * 64 + h * 16;
#pragma unroll
  for (int d = 0; d < 16; ++d) orow[d] = oa[d];
}

// ---------------------------------------------------------------------------
// Encoder MHA core: L=16, heads=16, hd=128. qkv: (16*bs, 6144) rows = l*bs+b.
// One block per (b,h). Phase1: 256 threads = (l,kl) pair dots. Phase2: att@V.
// ---------------------------------------------------------------------------
__global__ void enc_attn_kernel(const float* __restrict__ qkv, float* __restrict__ o, int bs) {
  int h = blockIdx.x & 15;
  int b = blockIdx.x >> 4;
  __shared__ float s[16][17];
  __shared__ float att[16][17];
  int l  = threadIdx.x >> 4;
  int kl = threadIdx.x & 15;
  const float scale = 0.08838834764831845f;  // 1/sqrt(128)
  const float* qr = qkv + ((long)l * bs + b) * 6144 + h * 128;
  const float* kr = qkv + ((long)kl * bs + b) * 6144 + 2048 + h * 128;
  float acc = 0.0f;
#pragma unroll 8
  for (int d = 0; d < 128; ++d) acc += qr[d] * kr[d];
  s[l][kl] = acc * scale;
  __syncthreads();
  if (threadIdx.x < 16) {
    int r = threadIdx.x;
    float mx = -3.4e38f;
#pragma unroll
    for (int j = 0; j < 16; ++j) mx = fmaxf(mx, s[r][j]);
    float sum = 0.0f;
#pragma unroll
    for (int j = 0; j < 16; ++j) { float e = __expf(s[r][j] - mx); att[r][j] = e; sum += e; }
    float inv = 1.0f / sum;
#pragma unroll
    for (int j = 0; j < 16; ++j) att[r][j] *= inv;
  }
  __syncthreads();
  int dg = kl;                         // 8-wide slice of hd per thread
  float oa[8] = {};
#pragma unroll
  for (int j = 0; j < 16; ++j) {
    float wgt = att[l][j];
    const float* vr = qkv + ((long)j * bs + b) * 6144 + 4096 + h * 128 + dg * 8;
#pragma unroll
    for (int u = 0; u < 8; ++u) oa[u] += wgt * vr[u];
  }
  float* orow = o + ((long)l * bs + b) * 2048 + h * 128 + dg * 8;
#pragma unroll
  for (int u = 0; u < 8; ++u) orow[u] = oa[u];
}

// ---------------------------------------------------------------------------
// out = LayerNorm(x + r) * w + b over rows of width D (D <= 2048). Block=256.
// ---------------------------------------------------------------------------
__global__ void add_ln_kernel(const float* __restrict__ x, const float* __restrict__ r,
                              const float* __restrict__ w, const float* __restrict__ b,
                              float* __restrict__ out, int D) {
  long row = blockIdx.x;
  const float* xr = x + row * D;
  const float* rr = r + row * D;
  float vloc[8];
  float s = 0.0f;
#pragma unroll
  for (int c = 0; c < 8; ++c) {
    int i = threadIdx.x + c * 256;
    float v = (i < D) ? (xr[i] + rr[i]) : 0.0f;
    vloc[c] = v;
    s += v;
  }
  __shared__ float red[256];
  red[threadIdx.x] = s;
  __syncthreads();
  for (int st = 128; st > 0; st >>= 1) {
    if (threadIdx.x < st) red[threadIdx.x] += red[threadIdx.x + st];
    __syncthreads();
  }
  float mu = red[0] / D;
  __syncthreads();
  float s2 = 0.0f;
#pragma unroll
  for (int c = 0; c < 8; ++c) {
    int i = threadIdx.x + c * 256;
    float d = (i < D) ? (vloc[c] - mu) : 0.0f;
    s2 += d * d;
  }
  red[threadIdx.x] = s2;
  __syncthreads();
  for (int st = 128; st > 0; st >>= 1) {
    if (threadIdx.x < st) red[threadIdx.x] += red[threadIdx.x + st];
    __syncthreads();
  }
  float inv = rsqrtf(red[0] / D + 1e-5f);
  float* orow = out + row * D;
#pragma unroll
  for (int c = 0; c < 8; ++c) {
    int i = threadIdx.x + c * 256;
    if (i < D) orow[i] = (vloc[c] - mu) * inv * w[i] + b[i];
  }
}

// ---------------------------------------------------------------------------
// Small fp32 FC for the final N=25 layer (not a multiple of 16).
// ---------------------------------------------------------------------------
__global__ void fc_small_kernel(const float* __restrict__ a, const float* __restrict__ w,
                                const float* __restrict__ b, float* __restrict__ out,
                                int M, int N, int K) {
  long idx = (long)blockIdx.x * blockDim.x + threadIdx.x;
  if (idx >= (long)M * N) return;
  int n = (int)(idx % N);
  long m = idx / N;
  const float* ar = a + m * (long)K;
  const float* wr = w + (long)n * K;
  float s = 0.0f;
  for (int k = 0; k < K; ++k) s += ar[k] * wr[k];
  out[idx] = s + b[n];
}

// ===========================================================================
extern "C" void kernel_launch(void* const* d_in, const int* in_sizes, int n_in,
                              void* d_out, int out_size, void* d_ws, size_t ws_size,
                              hipStream_t stream) {
  const float* t         = (const float*)d_in[0];
  const float* conv1_w   = (const float*)d_in[1];
  const float* conv1_b   = (const float*)d_in[2];
  const float* conv2_w   = (const float*)d_in[3];
  const float* conv2_b   = (const float*)d_in[4];
  const float* expand_w  = (const float*)d_in[5];
  const float* expand_b  = (const float*)d_in[6];
  const float* mha_wqkv  = (const float*)d_in[7];
  const float* mha_bqkv  = (const float*)d_in[8];
  const float* mha_wo    = (const float*)d_in[9];
  const float* mha_bo    = (const float*)d_in[10];
  const float* ln1_w     = (const float*)d_in[11];
  const float* ln1_b     = (const float*)d_in[12];
  const float* enc_wqkv  = (const float*)d_in[13];
  const float* enc_bqkv  = (const float*)d_in[14];
  const float* enc_wo    = (const float*)d_in[15];
  const float* enc_bo    = (const float*)d_in[16];
  const float* enc_ln1_w = (const float*)d_in[17];
  const float* enc_ln1_b = (const float*)d_in[18];
  const float* enc_w1    = (const float*)d_in[19];
  const float* enc_b1    = (const float*)d_in[20];
  const float* enc_w2    = (const float*)d_in[21];
  const float* enc_b2    = (const float*)d_in[22];
  const float* enc_ln2_w = (const float*)d_in[23];
  const float* enc_ln2_b = (const float*)d_in[24];
  const float* f1_w      = (const float*)d_in[25];
  const float* f1_b      = (const float*)d_in[26];
  const float* f2_w      = (const float*)d_in[27];
  const float* f2_b      = (const float*)d_in[28];
  const float* f3_w      = (const float*)d_in[29];
  const float* f3_b      = (const float*)d_in[30];
  (void)n_in; (void)out_size; (void)ws_size;

  const int  bs   = in_sizes[0] / (128 * 128);  // 256
  const long NIMG = (long)bs * 16;              // 4096 tile-images == encoder tokens
  const long RE   = NIMG * 32;                  // 131072 rows of 64-dim features
  const long TOK  = NIMG;                       // 16*bs encoder tokens
  const long Mb   = 32 * (long)bs;              // rows per group in grouped MHA

  char* Wb = (char*)d_ws;
  auto al = [](size_t v) { return (v + 255) & ~(size_t)255; };

  // ---- workspace regions (overlaid by lifetime) ----
  size_t szA = al((size_t)NIMG * 32 * 225 * 4);                 // u1 / qkv_g / qkv_enc
  { size_t q = al((size_t)RE * 192 * 4); if (q > szA) szA = q; }
  size_t sz33  = al((size_t)RE * 64 * 4);                       // 33.6MB class
  size_t szPck = al((size_t)(RE / 16) * 2 * 1024);              // largest activation pack
  size_t offA = 0;
  size_t offB = offA + szA;    // u2 -> o_g -> o_enc -> hbuf
  size_t offC = offB + sz33;   // rotating activation pack
  size_t offD = offC + szPck;  // uexp -> ff
  size_t offE = offD + sz33;   // attnout -> a_buf -> g1/g2
  size_t offF = offE + sz33;   // x_enc -> x2
  size_t offG = offF + sz33;   // x1
  size_t h = offG + sz33;      // persistent packed weights
  auto halloc = [&](size_t bytes) { size_t o = h; h = al(h + bytes); return o; };
  size_t off_expand_wp = halloc((64 / 16) * 2 * 1024);
  size_t off_mqkv_wp   = halloc((16 * 192 / 16) * 2 * 1024);
  size_t off_mwo_wp    = halloc((16 * 64 / 16) * 2 * 1024);
  size_t off_Wqkvp     = halloc((size_t)(6144 / 16) * 64 * 1024);
  size_t off_Wop       = halloc((size_t)(2048 / 16) * 64 * 1024);
  size_t off_W1p       = halloc((size_t)(2048 / 16) * 64 * 1024);
  size_t off_W2p       = halloc((size_t)(2048 / 16) * 64 * 1024);
  size_t off_F1p       = halloc((size_t)(512 / 16) * 1024 * 1024);
  size_t off_F2p       = halloc((size_t)(128 / 16) * 16 * 1024);

  auto PACK = [&](const float* X, size_t poff, long R, int K) {
    long nRT = (R + 15) / 16, nKT = (K + 31) / 32;
    long total = nRT * nKT * 32;
    unsigned blocks = (unsigned)((total + 255) / 256);
    pack_f16_kernel<<<dim3(blocks), dim3(256), 0, stream>>>(X, (_Float16*)(Wb + poff), R, K);
  };
  auto GEMM = [&](size_t aoff, size_t woff, const float* bias, float* C,
                  int M, int N, int K, int batches,
                  long aBS, long wBS, long bBS, long cBS, int relu) {
    long ntiles = (long)(M / 64) * (N / 32);    // 64x32 macro-tiles per wave
    dim3 grid((unsigned)((ntiles + WAVES_PER_BLOCK - 1) / WAVES_PER_BLOCK), batches);
    gemm_wmma_f16<<<grid, dim3(256), 0, stream>>>(
        (const _Float16*)(Wb + aoff), (const _Float16*)(Wb + woff), bias, C,
        M, N, K, aBS, wBS, bBS, cBS, relu);
  };

  // ---- weight packing (every call; deterministic) ----
  PACK(expand_w, off_expand_wp, 64, 36);
  PACK(mha_wqkv, off_mqkv_wp, 16 * 192, 64);
  PACK(mha_wo,   off_mwo_wp,  16 * 64, 64);
  PACK(enc_wqkv, off_Wqkvp, 6144, 2048);
  PACK(enc_wo,   off_Wop,   2048, 2048);
  PACK(enc_w1,   off_W1p,   2048, 2048);
  PACK(enc_w2,   off_W2p,   2048, 2048);
  PACK(f1_w,     off_F1p,    512, 32768);
  PACK(f2_w,     off_F2p,    128, 512);

  // ---- stage 1: tile + conv1 + pool ----
  float* u1 = (float*)(Wb + offA);
  {
    long total = NIMG * 32 * 225;
    conv1_pool_kernel<<<dim3((unsigned)((total + 255) / 256)), dim3(256), 0, stream>>>(
        t, conv1_w, conv1_b, u1, bs);
  }
  // ---- stage 2: conv2 + pool -> (NIMG,32,36) ----
  float* u2 = (float*)(Wb + offB);
  {
    long total = NIMG * 32 * 36;
    conv2_pool_kernel<<<dim3((unsigned)((total + 255) / 256)), dim3(256), 0, stream>>>(
        u1, conv2_w, conv2_b, u2, NIMG);
  }
  // ---- stage 3: expand (WMMA, K padded 36->64) + ReLU ----
  PACK(u2, offC, RE, 36);
  float* uexp = (float*)(Wb + offD);
  GEMM(offC, off_expand_wp, expand_b, uexp, (int)RE, 64, 36, 1, 0, 0, 0, 0, 1);

  // ---- stage 4: grouped MHA (16 groups batched WMMA QKV + VALU softmax + WMMA O) ----
  PACK(uexp, offC, RE, 64);
  float* qkv_g = (float*)(Wb + offA);
  GEMM(offC, off_mqkv_wp, mha_bqkv, qkv_g, (int)Mb, 192, 64, 16,
       (Mb / 16) * 2 * 512, (192 / 16) * 2 * 512, 192, Mb * 192, 0);
  float* o_g = (float*)(Wb + offB);
  grouped_attn_kernel<<<dim3((unsigned)(16 * bs * 4)), dim3(32), 0, stream>>>(qkv_g, o_g, bs);
  PACK(o_g, offC, RE, 64);
  float* attnout = (float*)(Wb + offE);
  GEMM(offC, off_mwo_wp, mha_bo, attnout, (int)Mb, 64, 64, 16,
       (Mb / 16) * 2 * 512, (64 / 16) * 2 * 512, 64, Mb * 64, 0);
  float* x_enc = (float*)(Wb + offF);   // LN over D=64; buffer doubles as (TOK,2048)
  add_ln_kernel<<<dim3((unsigned)RE), dim3(256), 0, stream>>>(uexp, attnout, ln1_w, ln1_b, x_enc, 64);

  // ---- stage 5: encoder self-attention ----
  PACK(x_enc, offC, TOK, 2048);
  float* qkv_e = (float*)(Wb + offA);
  GEMM(offC, off_Wqkvp, enc_bqkv, qkv_e, (int)TOK, 6144, 2048, 1, 0, 0, 0, 0, 0);
  float* o_e = (float*)(Wb + offB);
  enc_attn_kernel<<<dim3((unsigned)(bs * 16)), dim3(256), 0, stream>>>(qkv_e, o_e, bs);
  PACK(o_e, offC, TOK, 2048);
  float* a_buf = (float*)(Wb + offE);
  GEMM(offC, off_Wop, enc_bo, a_buf, (int)TOK, 2048, 2048, 1, 0, 0, 0, 0, 0);
  float* x1 = (float*)(Wb + offG);
  add_ln_kernel<<<dim3((unsigned)TOK), dim3(256), 0, stream>>>(x_enc, a_buf, enc_ln1_w, enc_ln1_b, x1, 2048);

  // ---- stage 6: encoder FFN ----
  PACK(x1, offC, TOK, 2048);
  float* hbuf = (float*)(Wb + offB);
  GEMM(offC, off_W1p, enc_b1, hbuf, (int)TOK, 2048, 2048, 1, 0, 0, 0, 0, 1);
  PACK(hbuf, offC, TOK, 2048);
  float* ff = (float*)(Wb + offD);
  GEMM(offC, off_W2p, enc_b2, ff, (int)TOK, 2048, 2048, 1, 0, 0, 0, 0, 0);
  float* x2 = (float*)(Wb + offF);
  add_ln_kernel<<<dim3((unsigned)TOK), dim3(256), 0, stream>>>(x1, ff, enc_ln2_w, enc_ln2_b, x2, 2048);

  // ---- stage 7: head f1 (WMMA, K=32768) -> f2 (WMMA) -> f3 (VALU, N=25) ----
  PACK(x2, offC, bs, 32768);            // raw reshape (bs, 16*2048) is contiguous
  float* g1 = (float*)(Wb + offE);
  GEMM(offC, off_F1p, f1_b, g1, bs, 512, 32768, 1, 0, 0, 0, 0, 1);
  PACK(g1, offC, bs, 512);
  float* g2 = (float*)(Wb + offE + (1 << 20));
  GEMM(offC, off_F2p, f2_b, g2, bs, 128, 512, 1, 0, 0, 0, 0, 1);
  fc_small_kernel<<<dim3((unsigned)((bs * 25 + 255) / 256)), dim3(256), 0, stream>>>(
      g2, f3_w, f3_b, (float*)d_out, bs, 25, 128);
}
